// Single_Translator_68650757259305
// MI455X (gfx1250) — matrix operand order repaired and checked by reading, not compiled, not run on hardware
//
#include <hip/hip_runtime.h>
#include <hip/hip_bf16.h>
#include <math.h>

typedef __bf16 bf16_t;
typedef __attribute__((ext_vector_type(16))) __bf16 v16bf;
typedef __attribute__((ext_vector_type(8)))  float  v8f;

#define BDIM  4096
#define DIN   4096
#define DE    1024
#define NEXP  6
#define EPS   1e-5f

__device__ __forceinline__ float leakyf(float v) { return v > 0.f ? v : 0.01f * v; }

__device__ __forceinline__ unsigned pack_bf16(float a, float b) {
    union { bf16_t h[2]; unsigned u; } u_;
    u_.h[0] = (bf16_t)a; u_.h[1] = (bf16_t)b;
    return u_.u;
}

// ---------------------------------------------------------------------------
// WMMA bf16 GEMM:  C[z] = act( A @ B[z] + bias[z] + Res )
// A fp32 [M,K] row-major (shared over z), B fp32 [K,N] row-major per z.
// 256 threads = 8 waves. Macro tile 128 x (64*NB), BK=32, LDS double-buffered.
// Wave layout 4x2: wave owns 2 m-subtiles x 2*NB n-subtiles -> 4*NB WMMAs and
// (2 + 2*NB) fragment loads per K-step, one barrier per K-step; global loads
// for step t+1 overlap the WMMAs of step t via register staging.
//   A frag map (16x32): lane = 16*(k/16) + m ; j = k%16
//   B frag map (32x16): lane = 16*(k/16) + n    ; j = k%16
//   C/D map: row = r + 8*(lane/16), col = lane%16
// ---------------------------------------------------------------------------
template <int NB, int HAS_BIAS, int HAS_RES, int ACT_RELU>
__global__ __launch_bounds__(256) void gemm_wmma_bf16(
    const float* __restrict__ A, int lda,
    const float* __restrict__ Bm, int ldb, long strideBz,
    float* __restrict__ C, int ldc, long strideCz,
    const float* __restrict__ bias, int strideBiasZ,
    const float* __restrict__ Res, int ldres,
    int K)
{
    __shared__ bf16_t sA[2][128 * 32];       // [buf][m_sub 0..7][lane][16]
    __shared__ bf16_t sB[2][NB * 64 * 32];   // [buf][n_sub][lane][16]

    const int t      = threadIdx.x;
    const int lane   = t & 31;
    const int w      = t >> 5;       // wave 0..7
    const int m_pair = w >> 1;       // 0..3 -> m subtiles {2*m_pair, +1}
    const int n_base = (w & 1) * 2 * NB;   // n subtiles n_base .. n_base+2*NB-1
    const int bm     = blockIdx.x * 128;
    const int bn     = blockIdx.y * (64 * NB);
    const int z      = blockIdx.z;

    const float* Bz = Bm + (long)z * strideBz;
    float*       Cz = C  + (long)z * strideCz;

    v8f acc[4 * NB];
    #pragma unroll
    for (int i = 0; i < 4 * NB; ++i) acc[i] = v8f{};

    // ---- register staging for global tiles
    float4 rA[2][2];               // 2 slots x 8 floats (8 consecutive k)
    float  rB[4 * NB][2];          // even-k pairs

    // A-slot geometry: slot = i*256+t -> (row 0..127, q 0..3); q = 8-k group.
    const int aRow0 = t >> 2;                  // slot i=0
    const int aQ0   = t & 3;
    const int aRow1 = (256 + t) >> 2;          // slot i=1
    const int aQ1   = t & 3;                   // (256+t)&3 == t&3

    auto loadG = [&](int k0) {
        const float* s0 = A + (long)(bm + aRow0) * lda + (k0 + aQ0 * 8);
        rA[0][0] = *(const float4*)s0;
        rA[0][1] = *(const float4*)(s0 + 4);
        const float* s1 = A + (long)(bm + aRow1) * lda + (k0 + aQ1 * 8);
        rA[1][0] = *(const float4*)s1;
        rA[1][1] = *(const float4*)(s1 + 4);
        #pragma unroll
        for (int i = 0; i < 4 * NB; ++i) {
            int p  = i * 256 + t;              // pair slot
            int kp = p / (64 * NB);            // 0..15
            int nL = p % (64 * NB);
            const float* src = Bz + (long)(k0 + kp * 2) * ldb + (bn + nL);
            rB[i][0] = src[0];
            rB[i][1] = src[ldb];
        }
    };

    auto storeL = [&](int buf) {
        #pragma unroll
        for (int i = 0; i < 2; ++i) {
            int row = i ? aRow1 : aRow0;
            int q   = i ? aQ1   : aQ0;
            uint4 pk;
            pk.x = pack_bf16(rA[i][0].x, rA[i][0].y);
            pk.y = pack_bf16(rA[i][0].z, rA[i][0].w);
            pk.z = pack_bf16(rA[i][1].x, rA[i][1].y);
            pk.w = pack_bf16(rA[i][1].z, rA[i][1].w);
            int laneA = ((q >> 1) << 4) + (row & 15);
            int jb    = (q & 1) << 3;
            *(uint4*)&sA[buf][((row >> 4) * 32 + laneA) * 16 + jb] = pk;
        }
        #pragma unroll
        for (int i = 0; i < 4 * NB; ++i) {
            int p  = i * 256 + t;
            int kp = p / (64 * NB);
            int nL = p % (64 * NB);
            int k  = kp * 2;
            int laneB = ((k >> 4) << 4) + (nL & 15);
            *(unsigned*)&sB[buf][((nL >> 4) * 32 + laneB) * 16 + (k & 15)] =
                pack_bf16(rB[i][0], rB[i][1]);
        }
    };

    auto compute = [&](int buf) {
        v16bf a0 = *(const v16bf*)&sA[buf][((m_pair * 2 + 0) * 32 + lane) * 16];
        v16bf a1 = *(const v16bf*)&sA[buf][((m_pair * 2 + 1) * 32 + lane) * 16];
        v16bf bfr[2 * NB];
        #pragma unroll
        for (int ni = 0; ni < 2 * NB; ++ni)
            bfr[ni] = *(const v16bf*)&sB[buf][((n_base + ni) * 32 + lane) * 16];
        #pragma unroll
        for (int ni = 0; ni < 2 * NB; ++ni) {
            acc[ni] = __builtin_amdgcn_wmma_f32_16x16x32_bf16(
                false, a0, false, bfr[ni], (short)0, acc[ni], false, false);
            acc[2 * NB + ni] = __builtin_amdgcn_wmma_f32_16x16x32_bf16(
                false, a1, false, bfr[ni], (short)0, acc[2 * NB + ni], false, false);
        }
    };

    const int T = K >> 5;
    loadG(0);
    storeL(0);
    for (int tt = 0; tt < T - 1; ++tt) {
        __syncthreads();
        loadG((tt + 1) << 5);
        compute(tt & 1);
        storeL((tt + 1) & 1);
    }
    __syncthreads();
    compute((T - 1) & 1);

    // ---- epilogue
    const int colHalf = lane & 15;
    const int rowHalf = (lane >> 4) * 8;
    #pragma unroll
    for (int mi = 0; mi < 2; ++mi) {
        const int rowBase = bm + (m_pair * 2 + mi) * 16 + rowHalf;
        #pragma unroll
        for (int ni = 0; ni < 2 * NB; ++ni) {
            const int col = bn + (n_base + ni) * 16 + colHalf;
            float bcol = 0.f;
            if (HAS_BIAS) bcol = bias[(long)z * strideBiasZ + col];
            const v8f a = acc[mi * 2 * NB + ni];
            #pragma unroll
            for (int r = 0; r < 8; ++r) {
                int row = rowBase + r;
                float v = a[r];
                if (HAS_BIAS) v += bcol;
                if (HAS_RES)  v += Res[(long)row * ldres + col];
                if (ACT_RELU) v = v > 0.f ? v : 0.f;
                Cz[(long)row * ldc + col] = v;
            }
        }
    }
}

// ---------------------------------------------------------------------------
// Gating tail: gate = softmax(G1 @ Wg2 + bg2), G1 [B,64], Wg2 [64,6]
// ---------------------------------------------------------------------------
__global__ __launch_bounds__(256) void gate_softmax_kernel(
    const float* __restrict__ G1, const float* __restrict__ Wg2,
    const float* __restrict__ bg2, float* __restrict__ gate, int rows)
{
    int b = blockIdx.x * blockDim.x + threadIdx.x;
    if (b >= rows) return;
    float acc[NEXP];
    #pragma unroll
    for (int e = 0; e < NEXP; ++e) acc[e] = bg2[e];
    const float* g1 = G1 + (long)b * 64;
    for (int k = 0; k < 64; ++k) {
        float g = g1[k];
        #pragma unroll
        for (int e = 0; e < NEXP; ++e) acc[e] += g * Wg2[k * NEXP + e];
    }
    float mx = acc[0];
    #pragma unroll
    for (int e = 1; e < NEXP; ++e) mx = fmaxf(mx, acc[e]);
    float sum = 0.f;
    #pragma unroll
    for (int e = 0; e < NEXP; ++e) { acc[e] = expf(acc[e] - mx); sum += acc[e]; }
    float inv = 1.f / sum;
    #pragma unroll
    for (int e = 0; e < NEXP; ++e) gate[(long)b * NEXP + e] = acc[e] * inv;
}

// ---------------------------------------------------------------------------
// Coalesced two-phase batch stats (biased var -> mean, rsqrt(var+eps))
// ---------------------------------------------------------------------------
__global__ __launch_bounds__(256) void bn_partial_kernel(
    const float* __restrict__ H, int ld, int rowsPerChunk, int cols,
    float* __restrict__ psum, float* __restrict__ psq)
{
    int c     = blockIdx.x * 256 + threadIdx.x;
    int chunk = blockIdx.y;
    long r0   = (long)chunk * rowsPerChunk;
    float a = 0.f, q = 0.f;
    for (int r = 0; r < rowsPerChunk; ++r) {
        float v = H[(r0 + r) * ld + c];
        a += v; q += v * v;
    }
    psum[(long)chunk * cols + c] = a;
    psq [(long)chunk * cols + c] = q;
}

__global__ __launch_bounds__(256) void bn_finalize_kernel(
    const float* __restrict__ psum, const float* __restrict__ psq,
    int cols, int chunks, float invRows,
    float* __restrict__ mean, float* __restrict__ rstd)
{
    int c = blockIdx.x * 256 + threadIdx.x;
    float a = 0.f, q = 0.f;
    for (int k = 0; k < chunks; ++k) {
        a += psum[(long)k * cols + c];
        q += psq [(long)k * cols + c];
    }
    float m   = a * invRows;
    float var = q * invRows - m * m;
    mean[c] = m;
    rstd[c] = rsqrtf(var + EPS);
}

// ---------------------------------------------------------------------------
// Fused BN-apply + leaky + gate-weighted expert combine for mu and d
// ---------------------------------------------------------------------------
__global__ __launch_bounds__(256) void combine_kernel(
    const float* __restrict__ h_mu, const float* __restrict__ h_d,
    const float* __restrict__ mean_mu, const float* __restrict__ rstd_mu,
    const float* __restrict__ g_mu, const float* __restrict__ b_mu,
    const float* __restrict__ mean_d, const float* __restrict__ rstd_d,
    const float* __restrict__ g_d, const float* __restrict__ b_d,
    const float* __restrict__ gate,
    float* __restrict__ mu_out, float* __restrict__ d_out_v)
{
    long idx = (long)blockIdx.x * blockDim.x + threadIdx.x;
    if (idx >= (long)BDIM * DE) return;
    int b = (int)(idx / DE);
    int o = (int)(idx % DE);
    const float* gr = gate + (long)b * NEXP;
    float am = 0.f, ad = 0.f;
    #pragma unroll
    for (int e = 0; e < NEXP; ++e) {
        int  c  = e * DE + o;
        float g = gr[e];
        float hm = h_mu[(long)b * (NEXP * DE) + c];
        float ym = g_mu[c] * (hm - mean_mu[c]) * rstd_mu[c] + b_mu[c];
        am += leakyf(ym) * g;
        float hd = h_d[(long)b * (NEXP * DE) + c];
        float yd = g_d[c] * (hd - mean_d[c]) * rstd_d[c] + b_d[c];
        ad += leakyf(yd) * g;
    }
    mu_out[idx]  = am;
    d_out_v[idx] = ad;
}

// ---------------------------------------------------------------------------
// Fused SE block: latent = mu * sigmoid(relu(mu @ Wse1) @ Wse2), one row/block
// ---------------------------------------------------------------------------
__global__ __launch_bounds__(256) void se_kernel(
    const float* __restrict__ mu, const float* __restrict__ Wse1,
    const float* __restrict__ Wse2, float* __restrict__ latent)
{
    __shared__ float row[DE];
    __shared__ float part[256 * 16];
    __shared__ float tv[16];
    int b = blockIdx.x;
    int t = threadIdx.x;
    #pragma unroll
    for (int i = 0; i < 4; ++i) row[t + i * 256] = mu[(long)b * DE + t + i * 256];
    __syncthreads();

    float acc[16];
    #pragma unroll
    for (int o = 0; o < 16; ++o) acc[o] = 0.f;
    #pragma unroll
    for (int i = 0; i < 4; ++i) {
        int k = t + i * 256;
        float v = row[k];
        #pragma unroll
        for (int o = 0; o < 16; ++o) acc[o] += v * Wse1[k * 16 + o];
    }
    #pragma unroll
    for (int o = 0; o < 16; ++o) part[t * 16 + o] = acc[o];
    __syncthreads();
    for (int s = 128; s > 0; s >>= 1) {
        if (t < s) {
            #pragma unroll
            for (int o = 0; o < 16; ++o) part[t * 16 + o] += part[(t + s) * 16 + o];
        }
        __syncthreads();
    }
    if (t < 16) tv[t] = fmaxf(part[t], 0.f);
    __syncthreads();

    #pragma unroll
    for (int i = 0; i < 4; ++i) {
        int o = t + i * 256;
        float s = 0.f;
        #pragma unroll
        for (int k = 0; k < 16; ++k) s += tv[k] * Wse2[k * DE + o];
        float sig = 1.f / (1.f + expf(-s));
        latent[(long)b * DE + o] = row[o] * sig;
    }
}

// ---------------------------------------------------------------------------
// Decoder BN apply + leaky
// ---------------------------------------------------------------------------
__global__ __launch_bounds__(256) void dec_apply_kernel(
    const float* __restrict__ pre, const float* __restrict__ mean,
    const float* __restrict__ rstd, const float* __restrict__ g,
    const float* __restrict__ bb, float* __restrict__ out)
{
    long idx = (long)blockIdx.x * blockDim.x + threadIdx.x;
    if (idx >= (long)BDIM * DIN) return;
    int c = (int)(idx % DIN);
    float v = g[c] * (pre[idx] - mean[c]) * rstd[c] + bb[c];
    out[idx] = leakyf(v);
}

// ---------------------------------------------------------------------------
extern "C" void kernel_launch(void* const* d_in, const int* in_sizes, int n_in,
                              void* d_out, int out_size, void* d_ws, size_t ws_size,
                              hipStream_t stream)
{
    const float* x     = (const float*)d_in[0];
    const float* We_mu = (const float*)d_in[1];
    const float* be_mu = (const float*)d_in[2];
    const float* g_mu  = (const float*)d_in[3];
    const float* b_mu  = (const float*)d_in[4];
    const float* We_d  = (const float*)d_in[5];
    const float* be_d  = (const float*)d_in[6];
    const float* g_d   = (const float*)d_in[7];
    const float* b_d   = (const float*)d_in[8];
    const float* Wg1   = (const float*)d_in[9];
    const float* bg1   = (const float*)d_in[10];
    const float* Wg2   = (const float*)d_in[11];
    const float* bg2   = (const float*)d_in[12];
    // Wq (13), Wk (14) are dead: softmax over a single key is identically 1.
    const float* Wv    = (const float*)d_in[15];
    const float* Wp    = (const float*)d_in[16];
    const float* bp    = (const float*)d_in[17];
    const float* Wse1  = (const float*)d_in[18];
    const float* Wse2  = (const float*)d_in[19];
    const float* Wd    = (const float*)d_in[20];
    const float* bd    = (const float*)d_in[21];
    const float* g_dec = (const float*)d_in[22];
    const float* b_dec = (const float*)d_in[23];

    float* out = (float*)d_out;
    float* muO = out + (long)BDIM * DIN;             // mu slot of d_out
    float* dO  = muO + (long)BDIM * DE;              // d slot of d_out

    // ---- workspace layout (floats), with aliasing ----
    float* ws = (float*)d_ws;
    const long HSZ  = (long)BDIM * NEXP * DE;        // 25165824
    float* h_mu   = ws;                              // [B,E,DE]
    float* h_d    = ws + HSZ;                        // [B,E,DE]
    float* pre    = h_mu;                            // aliases h_mu after combine
    float* v      = h_d;                             // aliases h_d after combine
    float* latent = h_d + (long)BDIM * DE;
    float* tail   = ws + 2 * HSZ;
    float* G1       = tail;                 tail += (long)BDIM * 64;
    float* gate     = tail;                 tail += (long)BDIM * NEXP;
    float* mean_mu  = tail;                 tail += NEXP * DE;
    float* rstd_mu  = tail;                 tail += NEXP * DE;
    float* mean_d   = tail;                 tail += NEXP * DE;
    float* rstd_d   = tail;                 tail += NEXP * DE;
    float* mean_dec = tail;                 tail += DIN;
    float* rstd_dec = tail;                 tail += DIN;
    float* psum     = tail;                 tail += 16 * NEXP * DE;
    float* psq      = tail;

    dim3 blk(256);
    const int CH = 16;                 // stat chunks
    const int RPC = BDIM / CH;         // rows per chunk

    // 1) expert GEMMs: h_mu/h_d[b,e,:] = x @ We[e] + be[e]
    {
        dim3 grid(BDIM / 128, DE / 128, NEXP);
        gemm_wmma_bf16<2, 1, 0, 0><<<grid, blk, 0, stream>>>(
            x, DIN, We_mu, DE, (long)DIN * DE,
            h_mu, NEXP * DE, (long)DE, be_mu, DE, nullptr, 0, DIN);
        gemm_wmma_bf16<2, 1, 0, 0><<<grid, blk, 0, stream>>>(
            x, DIN, We_d, DE, (long)DIN * DE,
            h_d, NEXP * DE, (long)DE, be_d, DE, nullptr, 0, DIN);
    }
    // 2) gating: G1 = relu(x @ Wg1 + bg1)   [B,64]  (BN=64 variant)
    {
        dim3 grid(BDIM / 128, 1, 1);
        gemm_wmma_bf16<1, 1, 0, 1><<<grid, blk, 0, stream>>>(
            x, DIN, Wg1, 64, 0,
            G1, 64, 0, bg1, 0, nullptr, 0, DIN);
    }
    // 3) gate = softmax(G1 @ Wg2 + bg2)
    gate_softmax_kernel<<<dim3((BDIM + 255) / 256), blk, 0, stream>>>(G1, Wg2, bg2, gate, BDIM);

    // 4) BN stats over batch for both expert tensors (coalesced two-phase)
    bn_partial_kernel<<<dim3(NEXP * DE / 256, CH), blk, 0, stream>>>(
        h_mu, NEXP * DE, RPC, NEXP * DE, psum, psq);
    bn_finalize_kernel<<<dim3(NEXP * DE / 256), blk, 0, stream>>>(
        psum, psq, NEXP * DE, CH, 1.f / (float)BDIM, mean_mu, rstd_mu);
    bn_partial_kernel<<<dim3(NEXP * DE / 256, CH), blk, 0, stream>>>(
        h_d, NEXP * DE, RPC, NEXP * DE, psum, psq);
    bn_finalize_kernel<<<dim3(NEXP * DE / 256), blk, 0, stream>>>(
        psum, psq, NEXP * DE, CH, 1.f / (float)BDIM, mean_d, rstd_d);

    // 5) BN apply + leaky + gate combine -> mu (d_out slot), d (d_out slot)
    combine_kernel<<<dim3((int)(((long)BDIM * DE + 255) / 256)), blk, 0, stream>>>(
        h_mu, h_d, mean_mu, rstd_mu, g_mu, b_mu,
        mean_d, rstd_d, g_d, b_d, gate, muO, dO);

    // 6) v = mu @ Wv   (attn == 1 => ao = v)
    {
        dim3 grid(BDIM / 128, DE / 128, 1);
        gemm_wmma_bf16<2, 0, 0, 0><<<grid, blk, 0, stream>>>(
            muO, DE, Wv, DE, 0,
            v, DE, 0, nullptr, 0, nullptr, 0, DE);
        // 7) mu = mu + v @ Wp + bp  (residual read+write on muO, disjoint/elem)
        gemm_wmma_bf16<2, 1, 1, 0><<<grid, blk, 0, stream>>>(
            v, DE, Wp, DE, 0,
            muO, DE, 0, bp, 0, muO, DE, DE);
    }
    // 8) SE gate: latent = mu * sigmoid(relu(mu@Wse1)@Wse2)
    se_kernel<<<dim3(BDIM), blk, 0, stream>>>(muO, Wse1, Wse2, latent);

    // 9) decoder GEMM: pre = latent @ Wd + bd   [B,DIN]
    {
        dim3 grid(BDIM / 128, DIN / 128, 1);
        gemm_wmma_bf16<2, 1, 0, 0><<<grid, blk, 0, stream>>>(
            latent, DE, Wd, DIN, 0,
            pre, DIN, 0, bd, 0, nullptr, 0, DE);
    }
    // 10) decoder BN stats + apply + leaky -> out
    bn_partial_kernel<<<dim3(DIN / 256, CH), blk, 0, stream>>>(
        pre, DIN, RPC, DIN, psum, psq);
    bn_finalize_kernel<<<dim3(DIN / 256), blk, 0, stream>>>(
        psum, psq, DIN, CH, 1.f / (float)BDIM, mean_dec, rstd_dec);
    dec_apply_kernel<<<dim3((int)(((long)BDIM * DIN + 255) / 256)), blk, 0, stream>>>(
        pre, mean_dec, rstd_dec, g_dec, b_dec, out);
}
